// HadamardRelationalAttention_55001351192727
// MI455X (gfx1250) — matrix operand order repaired and checked by reading, not compiled, not run on hardware
//
#include <hip/hip_runtime.h>
#include <hip/hip_bf16.h>

// Shapes (fixed by the reference): B=2, L=2048, DM=1024, H=16, D=64.
#define Lq   2048
#define DMq  1024
#define Hq   16
#define Dq   64
#define NTOK 4096   // B*L

typedef __bf16 v16bf __attribute__((ext_vector_type(16)));
typedef float  v8f   __attribute__((ext_vector_type(8)));
typedef unsigned int u32x4 __attribute__((ext_vector_type(4)));
typedef unsigned short u16;

union Frag32 { v16bf bf; u32x4 q[2]; u16 us[16]; };
union Pack8  { u32x4 q;  u16 us[8]; };

static __device__ __forceinline__ u16 f2bf(float f) {
  unsigned u = __float_as_uint(f);
  u += 0x7FFFu + ((u >> 16) & 1u);   // round-to-nearest-even
  return (u16)(u >> 16);
}

// B-fragment (16-bit, 32x16): lane = column (lane&15); lane>>4 picks K-half.
// Caller passes p already offset by (lane>>4)*16 K-elements; 32B contiguous.
static __device__ __forceinline__ v16bf ld_bfrag(const u16* p) {
  Frag32 f;
  f.q[0] = *reinterpret_cast<const u32x4*>(p);
  f.q[1] = *reinterpret_cast<const u32x4*>(p + 8);
  return f.bf;
}

// A-fragment (16-bit, 16x32): lane = row (lane&15); half = lane>>4.
// VGPR0-3 hold K = half*16+0..7, VGPR4-7 hold K = half*16+8..15.
static __device__ __forceinline__ v16bf ld_afrag(const u16* rowp, int hi) {
  Frag32 f;
  f.q[0] = *reinterpret_cast<const u32x4*>(rowp + hi * 16);
  f.q[1] = *reinterpret_cast<const u32x4*>(rowp + hi * 16 + 8);
  return f.bf;
}

#define WMMA_BF16(A,B,C) \
  __builtin_amdgcn_wmma_f32_16x16x32_bf16(false,(A),false,(B),(short)0,(C),false,false)

static __device__ __forceinline__ float rmax16(float x) {
  x = fmaxf(x, __shfl_xor(x, 1, 32));
  x = fmaxf(x, __shfl_xor(x, 2, 32));
  x = fmaxf(x, __shfl_xor(x, 4, 32));
  x = fmaxf(x, __shfl_xor(x, 8, 32));
  return x;
}
static __device__ __forceinline__ float rsum16(float x) {
  x += __shfl_xor(x, 1, 32);
  x += __shfl_xor(x, 2, 32);
  x += __shfl_xor(x, 4, 32);
  x += __shfl_xor(x, 8, 32);
  return x;
}

// ---------------- conversion kernels ----------------

__global__ void __launch_bounds__(256) cvt_x_kernel(const float* __restrict__ x,
                                                    u16* __restrict__ xb) {
  int i = blockIdx.x * 256 + threadIdx.x;     // exact: NTOK*DM / 256 blocks
  xb[i] = f2bf(x[i]);
}

// wT[mat][col][k] = bf16(w[mat][k][col]) ; 5 matrices of 1024x1024
__global__ void __launch_bounds__(256) cvt_wT_kernel(
    const float* __restrict__ w0, const float* __restrict__ w1,
    const float* __restrict__ w2, const float* __restrict__ w3,
    const float* __restrict__ w4, u16* __restrict__ wT) {
  int i = blockIdx.x * 256 + threadIdx.x;     // exact: 5*1024*1024 / 256 blocks
  int mat = i >> 20;
  int r = i & 1048575;
  int k = r >> 10, c = r & 1023;
  const float* w = (mat == 0) ? w0 : (mat == 1) ? w1 : (mat == 2) ? w2
                 : (mat == 3) ? w3 : w4;
  wT[(size_t)mat * 1048576 + (size_t)c * 1024 + k] = f2bf(w[r]);
}

// ---------------- fused projection GEMM ----------------
// One wave computes a 16x64 tile of Y = xb(4096x1024) @ W(1024x1024).
// blockIdx.y selects the weight matrix:
//   0 -> q  (bf16, token-major)     1 -> k   (bf16, token-major)
//   2 -> qr (f32,  token-major)     3 -> krT (bf16, [b,h,d,m] transposed)
__global__ void __launch_bounds__(256) proj_gemm_kernel(
    const u16* __restrict__ xb, const u16* __restrict__ wT,
    u16* __restrict__ qb, u16* __restrict__ kb,
    float* __restrict__ qr, u16* __restrict__ krT) {
  const int lane = threadIdx.x & 31;
  const int lo = lane & 15, hi = lane >> 4;
  const int wt = blockIdx.x * 8 + (threadIdx.x >> 5);  // 0..4095 wave tiles
  const int mat = blockIdx.y;
  const int row0 = (wt >> 4) * 16;
  const int col0 = (wt & 15) * 64;
  const u16* wTm = wT + (size_t)mat * 1048576;

  v8f acc[4] = {};
  const u16* arow = xb + (size_t)(row0 + lo) * DMq;
  for (int ks = 0; ks < 32; ++ks) {
    v16bf a = ld_afrag(arow + ks * 32, hi);
#pragma unroll
    for (int n = 0; n < 4; ++n) {
      v16bf bf = ld_bfrag(wTm + (size_t)(col0 + n * 16 + lo) * 1024 + ks * 32 + hi * 16);
      acc[n] = WMMA_BF16(a, bf, acc[n]);
    }
  }

  if (mat <= 1) {
    u16* y = (mat == 0) ? qb : kb;
#pragma unroll
    for (int n = 0; n < 4; ++n)
#pragma unroll
      for (int v = 0; v < 8; ++v)
        y[(size_t)(row0 + v + 8 * hi) * DMq + col0 + n * 16 + lo] = f2bf(acc[n][v]);
  } else if (mat == 2) {
#pragma unroll
    for (int n = 0; n < 4; ++n)
#pragma unroll
      for (int v = 0; v < 8; ++v)
        qr[(size_t)(row0 + v + 8 * hi) * DMq + col0 + n * 16 + lo] = acc[n][v];
  } else {
    // krT[b][h][d][m], m contiguous: 8 consecutive tokens -> one 16B store.
#pragma unroll
    for (int n = 0; n < 4; ++n) {
      int col = col0 + n * 16 + lo;
      int h = col >> 6, d = col & 63;
      int tok = row0 + 8 * hi;
      int bb = tok >> 11, m = tok & 2047;
      Pack8 p;
#pragma unroll
      for (int v = 0; v < 8; ++v) p.us[v] = f2bf(acc[n][v]);
      *reinterpret_cast<u32x4*>(krT + ((size_t)(bb * Hq + h) * Dq + d) * Lq + m) = p.q;
    }
  }
}

// ---------------- flash attention over relational keys ----------------
// One wave per (b, h, 16-query tile). Streams 32 keys per iteration:
//   S = Q K^T (4 WMMAs), online softmax, O += P @ Kr (4 WMMAs).
__global__ void __launch_bounds__(256) attn_kernel(
    const u16* __restrict__ qb, const u16* __restrict__ kb,
    const u16* __restrict__ krT, const float* __restrict__ qr,
    u16* __restrict__ had) {
  __shared__ alignas(16) u16 pbuf[8][16][32];   // per-wave P transpose buffer

  const int lane = threadIdx.x & 31;
  const int lo = lane & 15, hi = lane >> 4;
  const int widx = threadIdx.x >> 5;
  const int wt = blockIdx.x * 8 + widx;         // 0..4095
  const int qt = wt & 127;
  const int h = (wt >> 7) & 15;
  const int b = wt >> 11;
  const int q0 = qt * 16;

  const u16* qbase = qb + (size_t)b * Lq * DMq + h * Dq;
  const u16* kbase = kb + (size_t)b * Lq * DMq + h * Dq;
  const u16* krb = krT + (size_t)(b * Hq + h) * Dq * Lq;   // [d][m]

  const u16* qrow = qbase + (size_t)(q0 + lo) * DMq;
  v16bf qf0 = ld_afrag(qrow, hi);        // d = 0..31
  v16bf qf1 = ld_afrag(qrow + 32, hi);   // d = 32..63

  float M[8], Ls[8];
  v8f O[4] = {};
#pragma unroll
  for (int v = 0; v < 8; ++v) { M[v] = -1e30f; Ls[v] = 0.0f; }

  const int nkb = (qt >> 1) + 1;   // causal: keys [0, q0+16)
  for (int kbi = 0; kbi < nkb; ++kbi) {
    const int key0 = kbi * 32;

    // scores: two 16x16 tiles, K-dim d=64 in two WMMA steps each
    v8f s0 = {}, s1 = {};
    {
      const u16* kp0 = kbase + (size_t)(key0 + lo) * DMq + hi * 16;
      s0 = WMMA_BF16(qf0, ld_bfrag(kp0), s0);
      s0 = WMMA_BF16(qf1, ld_bfrag(kp0 + 32), s0);
      const u16* kp1 = kbase + (size_t)(key0 + 16 + lo) * DMq + hi * 16;
      s1 = WMMA_BF16(qf0, ld_bfrag(kp1), s1);
      s1 = WMMA_BF16(qf1, ld_bfrag(kp1 + 32), s1);
    }

    // online softmax (rows live on fixed 16-lane halves -> shfl_xor reductions)
#pragma unroll
    for (int v = 0; v < 8; ++v) {
      const int qi = q0 + v + 8 * hi;
      float a = s0[v] * 0.125f; if (key0 + lo > qi)      a = -1e30f;
      float c = s1[v] * 0.125f; if (key0 + 16 + lo > qi) c = -1e30f;
      float mnew = fmaxf(M[v], rmax16(fmaxf(a, c)));
      float alpha = __expf(M[v] - mnew);
      float p0 = __expf(a - mnew), p1 = __expf(c - mnew);
      Ls[v] = Ls[v] * alpha + rsum16(p0 + p1);
      M[v] = mnew;
      O[0][v] *= alpha; O[1][v] *= alpha; O[2][v] *= alpha; O[3][v] *= alpha;
      pbuf[widx][v + 8 * hi][lo] = f2bf(p0);
      pbuf[widx][v + 8 * hi][16 + lo] = f2bf(p1);
    }

    // re-fetch P as an A-fragment (per-wave LDS; DS ops are in-order per wave)
    Frag32 pf;
    const u16* prow = &pbuf[widx][lo][0];
    pf.q[0] = *reinterpret_cast<const u32x4*>(prow + hi * 16);
    pf.q[1] = *reinterpret_cast<const u32x4*>(prow + hi * 16 + 8);

    // O(16x64) += P(16x32) @ Kr(32x64); krT gives contiguous keys per lane
#pragma unroll
    for (int n = 0; n < 4; ++n) {
      const u16* kp = krb + (size_t)(n * 16 + lo) * Lq + key0 + hi * 16;
      O[n] = WMMA_BF16(pf.bf, ld_bfrag(kp), O[n]);
    }
  }

  // normalize, Hadamard with qr (f32), store bf16 token-major
  const float* qrb = qr + (size_t)b * Lq * DMq + h * Dq;
#pragma unroll
  for (int n = 0; n < 4; ++n)
#pragma unroll
    for (int v = 0; v < 8; ++v) {
      const int tok = q0 + v + 8 * hi;
      const int col = n * 16 + lo;
      float o = O[n][v] / Ls[v];
      float qv = qrb[(size_t)tok * DMq + col];
      had[((size_t)(b * Lq + tok)) * DMq + h * Dq + col] = f2bf(o * qv);
    }
}

// ---------------- output projection ----------------
__global__ void __launch_bounds__(256) out_gemm_kernel(
    const u16* __restrict__ had, const u16* __restrict__ wTo,
    float* __restrict__ out) {
  const int lane = threadIdx.x & 31;
  const int lo = lane & 15, hi = lane >> 4;
  const int wt = blockIdx.x * 8 + (threadIdx.x >> 5);
  const int row0 = (wt >> 4) * 16;
  const int col0 = (wt & 15) * 64;

  v8f acc[4] = {};
  const u16* arow = had + (size_t)(row0 + lo) * DMq;
  for (int ks = 0; ks < 32; ++ks) {
    v16bf a = ld_afrag(arow + ks * 32, hi);
#pragma unroll
    for (int n = 0; n < 4; ++n) {
      v16bf bf = ld_bfrag(wTo + (size_t)(col0 + n * 16 + lo) * 1024 + ks * 32 + hi * 16);
      acc[n] = WMMA_BF16(a, bf, acc[n]);
    }
  }
#pragma unroll
  for (int n = 0; n < 4; ++n)
#pragma unroll
    for (int v = 0; v < 8; ++v)
      out[(size_t)(row0 + v + 8 * hi) * DMq + col0 + n * 16 + lo] = acc[n][v];
}

extern "C" void kernel_launch(void* const* d_in, const int* in_sizes, int n_in,
                              void* d_out, int out_size, void* d_ws, size_t ws_size,
                              hipStream_t stream) {
  (void)in_sizes; (void)n_in; (void)out_size; (void)ws_size;
  const float* x   = (const float*)d_in[0];
  const float* wqa = (const float*)d_in[1];
  const float* wka = (const float*)d_in[2];
  const float* wqr = (const float*)d_in[3];
  const float* wkr = (const float*)d_in[4];
  const float* wo  = (const float*)d_in[5];

  char* ws = (char*)d_ws;
  const size_t MB = 1u << 20;
  u16*   xb  = (u16*)(ws);                 //  8 MiB bf16 x
  u16*   wT  = (u16*)(ws + 8 * MB);        // 10 MiB: 5 transposed bf16 weights
  u16*   qb  = (u16*)(ws + 18 * MB);       //  8 MiB
  u16*   kb  = (u16*)(ws + 26 * MB);       //  8 MiB
  float* qrf = (float*)(ws + 34 * MB);     // 16 MiB f32
  u16*   krT = (u16*)(ws + 50 * MB);       //  8 MiB [b,h,d,m]
  u16*   had = (u16*)(ws + 58 * MB);       //  8 MiB  (total 66 MiB)

  cvt_x_kernel<<<(NTOK * DMq) / 256, 256, 0, stream>>>(x, xb);
  cvt_wT_kernel<<<(5 * 1024 * 1024) / 256, 256, 0, stream>>>(wqa, wka, wqr, wkr, wo, wT);
  proj_gemm_kernel<<<dim3(512, 4), 256, 0, stream>>>(xb, wT, qb, kb, qrf, krT);
  attn_kernel<<<512, 256, 0, stream>>>(qb, kb, krT, qrf, had);
  out_gemm_kernel<<<512, 256, 0, stream>>>(had, wT + 4 * 1048576, (float*)d_out);
}
